// VariationalLSTM_67190468379273
// MI455X (gfx1250) — matrix operand order repaired and checked by reading, not compiled, not run on hardware
//
#include <hip/hip_runtime.h>
#include <stdint.h>

typedef __attribute__((ext_vector_type(16))) _Float16 v16h;
typedef __attribute__((ext_vector_type(8)))  _Float16 v8h;
typedef __attribute__((ext_vector_type(8)))  float    v8f;

#define BATCH    32768
#define SEQ      8
#define HD       128
#define PRED     12
#define BT       64          // batch rows per block
#define NTHREADS 128         // 4 wave32, one wave per 16 rows

// ---------------- LDS layout (bytes) ----------------
#define OFF_WHH0 0u          // 512x128 f16 = 131072 (decoder reuses as sampled w_hid)
#define OFF_WHH1 131072u     // 512x128 f16 = 131072
#define OFF_H0   262144u     // 64x128 f16  = 16384
#define OFF_H1   278528u     // 64x128 f16  = 16384
#define OFF_B0   294912u     // 512 f32     = 2048
#define OFF_B1   296960u     // 512 f32     = 2048
#define OFF_WI0  299008u     // 512x2 f32   = 4096
#define OFF_WIND 303104u     // 512x2 f32   = 4096 (sampled w_in)
#define OFF_BD   307200u     // 512 f32     = 2048 (sampled b_in+b_hid)
#define OFF_WOUT 309248u     // 4x128 f32   = 2048 (sampled w_out)
#define OFF_BOUT 311296u     // 4 f32       = 16
#define OFF_XT   311312u     // 64x2 f32    = 512
#define LDS_BYTES 311840u    // < 320 KB

// ---------------- workspace layout (bytes) ----------------
#define WS_WHID  0u                      // 12 * 65536 f16 = 1572864
#define WS_WIN   1572864u                // 12 * 1024 f32  = 49152
#define WS_BD    1622016u                // 12 * 512 f32   = 24576
#define WS_WOUT  1646592u                // 12 * 512 f32   = 24576
#define WS_BOUT  1671168u                // 12 * 4 f32     = 192

// ---------------- helpers ----------------
__device__ __forceinline__ float sigm(float x) { return 1.0f / (1.0f + __expf(-x)); }
__device__ __forceinline__ float tanh_(float x) {
  x = fminf(fmaxf(x, -15.0f), 15.0f);
  float e = __expf(2.0f * x);
  return (e - 1.0f) / (e + 1.0f);
}
__device__ __forceinline__ float softplus_(float x) {
  float ax = fabsf(x);
  return fmaxf(x, 0.0f) + __logf(1.0f + __expf(-ax));
}
__device__ __forceinline__ uint32_t lb32(uint32_t x) {
  x ^= x >> 16; x *= 0x7feb352du;
  x ^= x >> 15; x *= 0x846ca68bu;
  x ^= x >> 16;
  return x;
}
__device__ __forceinline__ float rng_normal(uint32_t idx, uint32_t salt) {
  uint32_t h1 = lb32(idx * 2u + 0x9e3779b9u * salt + 0x85ebca6bu);
  uint32_t h2 = lb32(idx * 2u + 1u + 0x9e3779b9u * salt + 0xc2b2ae35u);
  float u1 = (float)(h1 >> 8) * (1.0f / 16777216.0f) + 1.0e-7f;
  float u2 = (float)(h2 >> 8) * (1.0f / 16777216.0f);
  return __fsqrt_rn(-2.0f * __logf(u1)) * __cosf(6.28318530718f * u2);
}
__device__ __forceinline__ float drop_scale(uint32_t b, uint32_t t, uint32_t k) {
  uint32_t h = lb32(((b * 8u + t) * 128u + k) ^ 0x3c6ef372u);
  return (h < 3865470566u) ? (1.0f / 0.9f) : 0.0f;   // keep-prob 0.9, inverted scaling
}

__device__ __forceinline__ v16h pack16(v8h lo, v8h hi) {
  v16h r;
#pragma unroll
  for (int i = 0; i < 8; i++) { r[i] = lo[i]; r[i + 8] = hi[i]; }
  return r;
}

// A fragment (16x32 f16): rows = wave's h tile rows (row-major, stride HD halves)
__device__ __forceinline__ v16h load_a(const _Float16* rows, int kk, int lane) {
  const int m = lane & 15;
  const bool hi = lane >= 16;
  const _Float16* p = rows + m * HD;
  v8h lo = *(const v8h*)(p + kk + (hi ? 8 : 0));
  v8h hh = *(const v8h*)(p + kk + (hi ? 24 : 16));
  return pack16(lo, hh);
}

// A fragment with inline dropout on the h0 activations
__device__ __forceinline__ v16h load_a_drop(const _Float16* rows, int kk, int lane,
                                            uint32_t bWave, uint32_t t) {
  const int m = lane & 15;
  const bool hi = lane >= 16;
  const uint32_t b = bWave + (uint32_t)m;
  const _Float16* p = rows + m * HD;
  const int k0 = kk + (hi ? 8 : 0);
  const int k1 = kk + (hi ? 24 : 16);
  v8h lo = *(const v8h*)(p + k0);
  v8h hh = *(const v8h*)(p + k1);
  v16h r;
#pragma unroll
  for (int i = 0; i < 8; i++) {
    r[i]     = (_Float16)((float)lo[i] * drop_scale(b, t, (uint32_t)(k0 + i)));
    r[i + 8] = (_Float16)((float)hh[i] * drop_scale(b, t, (uint32_t)(k1 + i)));
  }
  return r;
}

// B fragment (32x16 f16) from an LDS weight matrix stored row-major (512 x 128)
__device__ __forceinline__ v16h load_b_lds(const _Float16* w, int col0, int kk, int lane) {
  const int n = col0 + (lane & 15);
  const int kb = kk + ((lane >= 16) ? 8 : 0);
  const _Float16* p = w + n * HD + kb;
  v8h lo = *(const v8h*)(p);
  v8h hh = *(const v8h*)(p + 16);
  return pack16(lo, hh);
}

// B fragment streamed from a global f32 weight matrix (row-major 512 x 128), cvt to f16
__device__ __forceinline__ v16h load_b_gf32(const float* w, int col0, int kk, int lane) {
  const int n = col0 + (lane & 15);
  const int kb = kk + ((lane >= 16) ? 8 : 0);
  const float4* p = (const float4*)(w + n * HD + kb);
  float4 f0 = p[0], f1 = p[1], f2 = p[4], f3 = p[5];
  v16h r;
  r[0] = (_Float16)f0.x;  r[1] = (_Float16)f0.y;  r[2] = (_Float16)f0.z;  r[3] = (_Float16)f0.w;
  r[4] = (_Float16)f1.x;  r[5] = (_Float16)f1.y;  r[6] = (_Float16)f1.z;  r[7] = (_Float16)f1.w;
  r[8] = (_Float16)f2.x;  r[9] = (_Float16)f2.y;  r[10] = (_Float16)f2.z; r[11] = (_Float16)f2.w;
  r[12] = (_Float16)f3.x; r[13] = (_Float16)f3.y; r[14] = (_Float16)f3.z; r[15] = (_Float16)f3.w;
  return r;
}

__device__ __forceinline__ v8f wmma_f16(v16h a, v16h b, v8f c) {
  return __builtin_amdgcn_wmma_f32_16x16x32_f16(false, a, false, b, (short)0, c, false, false);
}

// ==================================================================
// Pre-sample the decoder's Bayesian weights once (12 blocks, 1 per step)
// ==================================================================
__global__ void __launch_bounds__(256)
sample_weights_kernel(const float* __restrict__ din_wmu, const float* __restrict__ din_wrho,
                      const float* __restrict__ din_bmu, const float* __restrict__ din_brho,
                      const float* __restrict__ dhid_wmu, const float* __restrict__ dhid_wrho,
                      const float* __restrict__ dhid_bmu, const float* __restrict__ dhid_brho,
                      const float* __restrict__ out_wmu, const float* __restrict__ out_wrho,
                      const float* __restrict__ out_bmu, const float* __restrict__ out_brho,
                      _Float16* __restrict__ whid_all, float* __restrict__ win_all,
                      float* __restrict__ bd_all, float* __restrict__ wout_all,
                      float* __restrict__ bout_all) {
  const int s = blockIdx.x;
  const int tid = threadIdx.x;
  for (int i = tid; i < 512 * HD; i += 256)
    whid_all[(size_t)s * 65536 + i] =
        (_Float16)(dhid_wmu[i] + softplus_(dhid_wrho[i]) * rng_normal((uint32_t)i, (uint32_t)(s * 8 + 2)));
  for (int i = tid; i < 1024; i += 256)
    win_all[s * 1024 + i] = din_wmu[i] + softplus_(din_wrho[i]) * rng_normal((uint32_t)i, (uint32_t)(s * 8 + 0));
  for (int i = tid; i < 512; i += 256)
    bd_all[s * 512 + i] = din_bmu[i]  + softplus_(din_brho[i])  * rng_normal((uint32_t)i, (uint32_t)(s * 8 + 1))
                        + dhid_bmu[i] + softplus_(dhid_brho[i]) * rng_normal((uint32_t)i, (uint32_t)(s * 8 + 3));
  for (int i = tid; i < 512; i += 256)
    wout_all[s * 512 + i] = out_wmu[i] + softplus_(out_wrho[i]) * rng_normal((uint32_t)i, (uint32_t)(s * 8 + 4));
  if (tid < 4)
    bout_all[s * 4 + tid] = out_bmu[tid] + softplus_(out_brho[tid]) * rng_normal((uint32_t)tid, (uint32_t)(s * 8 + 5));
}

// ==================================================================
// Fused kernel: enc layer0 + dropout + enc layer1 + Bayesian decoder
// ==================================================================
__global__ void __launch_bounds__(NTHREADS)
vlstm_fused_kernel(const float* __restrict__ obs,
                   const float* __restrict__ wih0g, const float* __restrict__ whh0g,
                   const float* __restrict__ bih0,  const float* __restrict__ bhh0,
                   const float* __restrict__ wih1g, const float* __restrict__ whh1g,
                   const float* __restrict__ bih1,  const float* __restrict__ bhh1,
                   const _Float16* __restrict__ whid_all, const float* __restrict__ win_all,
                   const float* __restrict__ bd_all, const float* __restrict__ wout_all,
                   const float* __restrict__ bout_all,
                   float* __restrict__ out_mu, float* __restrict__ out_ls) {
  extern __shared__ char smem[];
  _Float16* WHH0 = (_Float16*)(smem + OFF_WHH0);   // decoder: sampled w_hid
  _Float16* WHH1 = (_Float16*)(smem + OFF_WHH1);
  _Float16* H0T  = (_Float16*)(smem + OFF_H0);
  _Float16* H1T  = (_Float16*)(smem + OFF_H1);
  float* B0   = (float*)(smem + OFF_B0);
  float* B1   = (float*)(smem + OFF_B1);
  float* WI0  = (float*)(smem + OFF_WI0);
  float* WIND = (float*)(smem + OFF_WIND);
  float* BD   = (float*)(smem + OFF_BD);
  float* WOUT = (float*)(smem + OFF_WOUT);
  float* BOUT = (float*)(smem + OFF_BOUT);
  float* XT   = (float*)(smem + OFF_XT);

  const int tid = threadIdx.x;
  const int lane = tid & 31;
  const int wave = tid >> 5;
  const int rowBase = wave * 16;          // first row of this wave's 16-row tile
  const bool hiHalf = (lane >= 16);
  const int nlan = lane & 15;
  const uint32_t bBlock = blockIdx.x * BT;
  const uint32_t bWave  = bBlock + (uint32_t)rowBase;

  // ---- cooperative fill: weights (f32->f16), biases, zero h tiles, last obs ----
  for (int i = tid; i < 512 * HD; i += NTHREADS) {
    WHH0[i] = (_Float16)whh0g[i];
    WHH1[i] = (_Float16)whh1g[i];
  }
  for (int i = tid; i < 512; i += NTHREADS) {
    B0[i] = bih0[i] + bhh0[i];
    B1[i] = bih1[i] + bhh1[i];
  }
  for (int i = tid; i < 1024; i += NTHREADS) WI0[i] = wih0g[i];
  for (int i = tid; i < BT * HD; i += NTHREADS) { H0T[i] = (_Float16)0.0f; H1T[i] = (_Float16)0.0f; }
  if (tid < BT) {
    const float* xp = obs + ((size_t)(bBlock + tid) * SEQ + (SEQ - 1)) * 2;
    XT[tid * 2 + 0] = xp[0];
    XT[tid * 2 + 1] = xp[1];
  }
  __syncthreads();

  // cell states live in D-fragment registers for the whole kernel
  float c0[8][8], c1[8][8];
#pragma unroll
  for (int nb = 0; nb < 8; nb++)
#pragma unroll
    for (int e = 0; e < 8; e++) { c0[nb][e] = 0.0f; c1[nb][e] = 0.0f; }

  const _Float16* h0row = H0T + rowBase * HD;
  const _Float16* h1row = H1T + rowBase * HD;

  // =========================== encoder ===========================
  for (int t = 0; t < SEQ; t++) {
    float x0[8], x1[8];
#pragma unroll
    for (int e = 0; e < 8; e++) {
      int m = hiHalf ? (8 + e) : e;
      const float* xp = obs + ((size_t)(bWave + m) * SEQ + t) * 2;
      x0[e] = xp[0]; x1[e] = xp[1];
    }

    // ---- layer 0: g = x@wih0^T + h0@whh0^T + b ----
    float hn[8][8];
    {
      v16h af[4];
#pragma unroll
      for (int q = 0; q < 4; q++) af[q] = load_a(h0row, q * 32, lane);   // hoisted: reused by all nb
#pragma unroll
      for (int nb = 0; nb < 8; nb++) {
        const int ncol = nb * 16 + nlan;
        v8f acc[4];
#pragma unroll
        for (int g = 0; g < 4; g++) {
          const int col = g * 128 + ncol;
          const float bias = B0[col];
          const float w0 = WI0[col * 2 + 0], w1 = WI0[col * 2 + 1];
#pragma unroll
          for (int e = 0; e < 8; e++) acc[g][e] = bias + x0[e] * w0 + x1[e] * w1;
        }
#pragma unroll
        for (int q = 0; q < 4; q++)
#pragma unroll
          for (int g = 0; g < 4; g++)
            acc[g] = wmma_f16(af[q], load_b_lds(WHH0, g * 128 + nb * 16, q * 32, lane), acc[g]);
#pragma unroll
        for (int e = 0; e < 8; e++) {
          float ig = sigm(acc[0][e]), fg = sigm(acc[1][e]);
          float gg = tanh_(acc[2][e]), og = sigm(acc[3][e]);
          float c = fg * c0[nb][e] + ig * gg;
          c0[nb][e] = c;
          hn[nb][e] = og * tanh_(c);
        }
      }
    }
#pragma unroll
    for (int nb = 0; nb < 8; nb++) {
      const int ncol = nb * 16 + nlan;
#pragma unroll
      for (int e = 0; e < 8; e++) {
        int m = hiHalf ? (8 + e) : e;
        H0T[(rowBase + m) * HD + ncol] = (_Float16)hn[nb][e];
      }
    }

    // ---- layer 1: g = drop(h0)@wih1^T + h1@whh1^T + b ----
    {
      v16h ad[4], a1[4];
#pragma unroll
      for (int q = 0; q < 4; q++) {
        ad[q] = load_a_drop(h0row, q * 32, lane, bWave, (uint32_t)t);    // dropout hashed once
        a1[q] = load_a(h1row, q * 32, lane);
      }
#pragma unroll
      for (int nb = 0; nb < 8; nb++) {
        const int ncol = nb * 16 + nlan;
        v8f acc[4];
#pragma unroll
        for (int g = 0; g < 4; g++) {
          const float bias = B1[g * 128 + ncol];
#pragma unroll
          for (int e = 0; e < 8; e++) acc[g][e] = bias;
        }
#pragma unroll
        for (int q = 0; q < 4; q++)
#pragma unroll
          for (int g = 0; g < 4; g++)
            acc[g] = wmma_f16(ad[q], load_b_gf32(wih1g, g * 128 + nb * 16, q * 32, lane), acc[g]);
#pragma unroll
        for (int q = 0; q < 4; q++)
#pragma unroll
          for (int g = 0; g < 4; g++)
            acc[g] = wmma_f16(a1[q], load_b_lds(WHH1, g * 128 + nb * 16, q * 32, lane), acc[g]);
#pragma unroll
        for (int e = 0; e < 8; e++) {
          float ig = sigm(acc[0][e]), fg = sigm(acc[1][e]);
          float gg = tanh_(acc[2][e]), og = sigm(acc[3][e]);
          float c = fg * c1[nb][e] + ig * gg;
          c1[nb][e] = c;
          hn[nb][e] = og * tanh_(c);
        }
      }
    }
#pragma unroll
    for (int nb = 0; nb < 8; nb++) {
      const int ncol = nb * 16 + nlan;
#pragma unroll
      for (int e = 0; e < 8; e++) {
        int m = hiHalf ? (8 + e) : e;
        H1T[(rowBase + m) * HD + ncol] = (_Float16)hn[nb][e];
      }
    }
  }

  // =========================== decoder ===========================
  for (int s = 0; s < PRED; s++) {
    __syncthreads();   // previous step's reads finished before overwriting samples
    // stage this step's pre-sampled weights from global (L2-resident) into LDS
    {
      const uint4* src = (const uint4*)(whid_all + (size_t)s * 65536);
      uint4* dst = (uint4*)WHH0;
      for (int i = tid; i < 8192; i += NTHREADS) dst[i] = src[i];
      for (int i = tid; i < 1024; i += NTHREADS) WIND[i] = win_all[s * 1024 + i];
      for (int i = tid; i < 512; i += NTHREADS) {
        BD[i]   = bd_all[s * 512 + i];
        WOUT[i] = wout_all[s * 512 + i];
      }
      if (tid < 4) BOUT[tid] = bout_all[s * 4 + tid];
    }
    __syncthreads();

    float x0[8], x1[8];
#pragma unroll
    for (int e = 0; e < 8; e++) {
      int m = hiHalf ? (8 + e) : e;
      x0[e] = XT[(rowBase + m) * 2 + 0];
      x1[e] = XT[(rowBase + m) * 2 + 1];
    }

    float hn[8][8];
    {
      v16h af[4];
#pragma unroll
      for (int q = 0; q < 4; q++) af[q] = load_a(h1row, q * 32, lane);
#pragma unroll
      for (int nb = 0; nb < 8; nb++) {
        const int ncol = nb * 16 + nlan;
        v8f acc[4];
#pragma unroll
        for (int g = 0; g < 4; g++) {
          const int col = g * 128 + ncol;
          const float bias = BD[col];
          const float w0 = WIND[col * 2 + 0], w1 = WIND[col * 2 + 1];
#pragma unroll
          for (int e = 0; e < 8; e++) acc[g][e] = bias + x0[e] * w0 + x1[e] * w1;
        }
#pragma unroll
        for (int q = 0; q < 4; q++)
#pragma unroll
          for (int g = 0; g < 4; g++)
            acc[g] = wmma_f16(af[q], load_b_lds(WHH0, g * 128 + nb * 16, q * 32, lane), acc[g]);
#pragma unroll
        for (int e = 0; e < 8; e++) {
          float ig = sigm(acc[0][e]), fg = sigm(acc[1][e]);
          float gg = tanh_(acc[2][e]), og = sigm(acc[3][e]);
          float c = fg * c1[nb][e] + ig * gg;
          c1[nb][e] = c;
          hn[nb][e] = og * tanh_(c);
        }
      }
    }
#pragma unroll
    for (int nb = 0; nb < 8; nb++) {
      const int ncol = nb * 16 + nlan;
#pragma unroll
      for (int e = 0; e < 8; e++) {
        int m = hiHalf ? (8 + e) : e;
        H1T[(rowBase + m) * HD + ncol] = (_Float16)hn[nb][e];
      }
    }

    // output projection: lanes 0-15 -> mu (rows), lanes 16-31 -> log_sigma (rows)
    {
      const int r = nlan;
      const int og = hiHalf ? 2 : 0;
      float o0 = BOUT[og], o1 = BOUT[og + 1];
      const _Float16* hr = H1T + (rowBase + r) * HD;
      for (int k = 0; k < HD; k++) {
        float hv = (float)hr[k];
        o0 += hv * WOUT[og * 128 + k];
        o1 += hv * WOUT[(og + 1) * 128 + k];
      }
      size_t ob = ((size_t)(bWave + r) * PRED + s) * 2;
      if (!hiHalf) {
        out_mu[ob + 0] = o0;
        out_mu[ob + 1] = o1;
        XT[(rowBase + r) * 2 + 0] = o0;   // x for next step = mu_xy
        XT[(rowBase + r) * 2 + 1] = o1;
      } else {
        out_ls[ob + 0] = fminf(fmaxf(o0, -4.0f), 2.0f);
        out_ls[ob + 1] = fminf(fmaxf(o1, -4.0f), 2.0f);
      }
    }
  }
}

// ================= KL reduction (tiny) =================
__global__ void kl_kernel(const float* dwm, const float* dwr, const float* dbm, const float* dbr,
                          const float* hwm, const float* hwr, const float* hbm, const float* hbr,
                          const float* owm, const float* owr, const float* obm, const float* obr,
                          float* out_kl) {
  __shared__ float red[256];
  float s = 0.0f;
  auto acc = [&](const float* mu, const float* rho, int n) {
    for (int i = threadIdx.x; i < n; i += 256) {
      float sd = softplus_(rho[i]);
      float m = mu[i];
      s += __logf(0.1f / sd) + (sd * sd + m * m) * 50.0f - 0.5f;   // 1/(2*0.1^2)=50
    }
  };
  acc(dwm, dwr, 1024); acc(dbm, dbr, 512);
  acc(hwm, hwr, 65536); acc(hbm, hbr, 512);
  acc(owm, owr, 512);  acc(obm, obr, 4);
  red[threadIdx.x] = s;
  __syncthreads();
  for (int st = 128; st > 0; st >>= 1) {
    if (threadIdx.x < (unsigned)st) red[threadIdx.x] += red[threadIdx.x + st];
    __syncthreads();
  }
  if (threadIdx.x == 0) out_kl[0] = red[0];
}

extern "C" void kernel_launch(void* const* d_in, const int* in_sizes, int n_in,
                              void* d_out, int out_size, void* d_ws, size_t ws_size,
                              hipStream_t stream) {
  (void)in_sizes; (void)n_in; (void)ws_size; (void)out_size;
  const float* obs       = (const float*)d_in[0];
  const float* wih0      = (const float*)d_in[1];
  const float* whh0      = (const float*)d_in[2];
  const float* bih0      = (const float*)d_in[3];
  const float* bhh0      = (const float*)d_in[4];
  const float* wih1      = (const float*)d_in[5];
  const float* whh1      = (const float*)d_in[6];
  const float* bih1      = (const float*)d_in[7];
  const float* bhh1      = (const float*)d_in[8];
  const float* din_wmu   = (const float*)d_in[9];
  const float* din_wrho  = (const float*)d_in[10];
  const float* din_bmu   = (const float*)d_in[11];
  const float* din_brho  = (const float*)d_in[12];
  const float* dhid_wmu  = (const float*)d_in[13];
  const float* dhid_wrho = (const float*)d_in[14];
  const float* dhid_bmu  = (const float*)d_in[15];
  const float* dhid_brho = (const float*)d_in[16];
  const float* out_wmu   = (const float*)d_in[17];
  const float* out_wrho  = (const float*)d_in[18];
  const float* out_bmu   = (const float*)d_in[19];
  const float* out_brho  = (const float*)d_in[20];

  float* out_mu = (float*)d_out;
  float* out_ls = out_mu + (size_t)BATCH * PRED * 2;
  float* out_kl = out_ls + (size_t)BATCH * PRED * 2;

  _Float16* ws_whid = (_Float16*)((char*)d_ws + WS_WHID);
  float*    ws_win  = (float*)((char*)d_ws + WS_WIN);
  float*    ws_bd   = (float*)((char*)d_ws + WS_BD);
  float*    ws_wout = (float*)((char*)d_ws + WS_WOUT);
  float*    ws_bout = (float*)((char*)d_ws + WS_BOUT);

  // 1) sample all decoder weights once (12 blocks, one per pred step)
  sample_weights_kernel<<<PRED, 256, 0, stream>>>(
      din_wmu, din_wrho, din_bmu, din_brho,
      dhid_wmu, dhid_wrho, dhid_bmu, dhid_brho,
      out_wmu, out_wrho, out_bmu, out_brho,
      ws_whid, ws_win, ws_bd, ws_wout, ws_bout);

  // 2) fused LSTM encoder + decoder
  (void)hipFuncSetAttribute(reinterpret_cast<const void*>(vlstm_fused_kernel),
                            hipFuncAttributeMaxDynamicSharedMemorySize, (int)LDS_BYTES);
  dim3 grid(BATCH / BT), block(NTHREADS);
  vlstm_fused_kernel<<<grid, block, LDS_BYTES, stream>>>(
      obs, wih0, whh0, bih0, bhh0, wih1, whh1, bih1, bhh1,
      ws_whid, ws_win, ws_bd, ws_wout, ws_bout,
      out_mu, out_ls);

  // 3) KL scalar
  kl_kernel<<<1, 256, 0, stream>>>(din_wmu, din_wrho, din_bmu, din_brho,
                                   dhid_wmu, dhid_wrho, dhid_bmu, dhid_brho,
                                   out_wmu, out_wrho, out_bmu, out_brho, out_kl);
}
